// CausalAlibiAttn_32040456028489
// MI455X (gfx1250) — matrix operand order repaired and checked by reading, not compiled, not run on hardware
//
#include <hip/hip_runtime.h>
#include <hip/hip_bf16.h>

// ---------------------------------------------------------------------------
// CausalAlibiAttn for MI455X (gfx1250): bf16 WMMA everywhere, f32 accumulate.
// B=2, S=2048, D=2048, H=16, KV=4, DH=128, n_rep=4.
// ---------------------------------------------------------------------------

#define B_ 2
#define S_ 2048
#define D_ 2048
#define H_ 16
#define KV_ 4
#define DH_ 128
#define KVD_ (KV_ * DH_)        // 512
#define ROWS_ ((size_t)B_ * S_) // 4096
#define SCALE_ 0.08838834764831845f

typedef __attribute__((ext_vector_type(16))) __bf16 v16bf;
typedef __attribute__((ext_vector_type(8)))  __bf16 v8bf;
typedef __attribute__((ext_vector_type(8)))  float  v8f;

// WMMA D = A(16x32 bf16) x B(32x16 bf16) + C(16x16 f32)
__device__ __forceinline__ v8f wmma_bf16(v16bf a, v16bf b, v8f c) {
  return __builtin_amdgcn_wmma_f32_16x16x32_bf16(
      /*neg_a=*/false, a, /*neg_b=*/false, b,
      /*c_mod=*/(short)0, c, /*reuse_a=*/false, /*reuse_b=*/false);
}

// A-matrix (16x32) per-lane layout: lanes 0-15 = rows, half = lane>>4.
// elements 0..7  = K[half*8 + 0..7], elements 8..15 = K[16 + half*8 + 0..7]
__device__ __forceinline__ v16bf load_a_tile(const __bf16* base, int half) {
  v8bf lo = *(const v8bf*)(base + half * 8);
  v8bf hi = *(const v8bf*)(base + 16 + half * 8);
  v16bf r;
#pragma unroll
  for (int i = 0; i < 8; ++i) { r[i] = lo[i]; r[i + 8] = hi[i]; }
  return r;
}

// B-matrix (32x16) per-lane layout: lane&15 = column, elements 0..7 =
// K[half*8 + 0..7], elements 8..15 = K[16 + half*8 + 0..7]. Two 16B loads keep 16B alignment.
__device__ __forceinline__ v16bf load_b_tile(const __bf16* base, int half) {
  v8bf lo = *(const v8bf*)(base + half * 8);
  v8bf hi = *(const v8bf*)(base + 16 + half * 8);
  v16bf r;
#pragma unroll
  for (int i = 0; i < 8; ++i) { r[i] = lo[i]; r[i + 8] = hi[i]; }
  return r;
}

__device__ __forceinline__ float rmax16(float v) {
  v = fmaxf(v, __shfl_xor(v, 8, 16));
  v = fmaxf(v, __shfl_xor(v, 4, 16));
  v = fmaxf(v, __shfl_xor(v, 2, 16));
  v = fmaxf(v, __shfl_xor(v, 1, 16));
  return v;
}
__device__ __forceinline__ float rsum16(float v) {
  v += __shfl_xor(v, 8, 16);
  v += __shfl_xor(v, 4, 16);
  v += __shfl_xor(v, 2, 16);
  v += __shfl_xor(v, 1, 16);
  return v;
}

// ---------------------------------------------------------------------------
__global__ void f32_to_bf16_kernel(const float* __restrict__ in,
                                   __bf16* __restrict__ out, size_t n) {
  size_t i = (size_t)blockIdx.x * blockDim.x + threadIdx.x;
  size_t stride = (size_t)gridDim.x * blockDim.x;
  for (; i < n; i += stride) out[i] = (__bf16)in[i];
}

// ---------------------------------------------------------------------------
// C(MxN) = A(MxK bf16, row major) * W^T (+ bias). W is (N,K) row-major, so
// B-matrix column n == W row n (contiguous).
// mode 0: f32 store  mode 1: bf16 store  mode 2: bf16 transposed (Vt layout).
// Block = 128 threads (4 waves). Block tile: 32 rows x 256 cols.
// Wave tile: 32x64 (2 M-tiles x 4 N-tiles), double-buffered operands.
__global__ void gemm_bf16_kernel(const __bf16* __restrict__ A,
                                 const __bf16* __restrict__ W,
                                 const float* __restrict__ bias,
                                 void* __restrict__ Cout,
                                 int M, int N, int K, int mode) {
  const int lane = threadIdx.x & 31;
  const int wave = threadIdx.x >> 5;
  const int half = lane >> 4;
  const int lr = lane & 15;
  const int m0 = blockIdx.x * 32;
  const int n0 = blockIdx.y * 256 + wave * 64;
  (void)M;

  v8f acc[2][4] = {};
  const __bf16* ar[2] = {A + (size_t)(m0 + lr) * K,
                         A + (size_t)(m0 + 16 + lr) * K};
  const __bf16* wb = W + (size_t)(n0 + lr) * K;

  // prologue: load k=0 operands
  v16bf a_c[2], b_c[4];
#pragma unroll
  for (int mt = 0; mt < 2; ++mt) a_c[mt] = load_a_tile(ar[mt], half);
#pragma unroll
  for (int t = 0; t < 4; ++t)
    b_c[t] = load_b_tile(wb + (size_t)(t * 16) * K, half);

  for (int k0 = 0; k0 < K; k0 += 32) {
    // issue next-k loads while current WMMAs execute
    const int kn = (k0 + 32 < K) ? k0 + 32 : k0;
    v16bf a_n[2], b_n[4];
#pragma unroll
    for (int mt = 0; mt < 2; ++mt) a_n[mt] = load_a_tile(ar[mt] + kn, half);
#pragma unroll
    for (int t = 0; t < 4; ++t)
      b_n[t] = load_b_tile(wb + (size_t)(t * 16) * K + kn, half);

#pragma unroll
    for (int mt = 0; mt < 2; ++mt)
#pragma unroll
      for (int t = 0; t < 4; ++t)
        acc[mt][t] = wmma_bf16(a_c[mt], b_c[t], acc[mt][t]);

#pragma unroll
    for (int mt = 0; mt < 2; ++mt) a_c[mt] = a_n[mt];
#pragma unroll
    for (int t = 0; t < 4; ++t) b_c[t] = b_n[t];
  }

#pragma unroll
  for (int t = 0; t < 4; ++t) {
    const int col = n0 + t * 16 + lr;
    const float bval = bias[col];
#pragma unroll
    for (int mt = 0; mt < 2; ++mt) {
#pragma unroll
      for (int j = 0; j < 8; ++j) {
        const int row = m0 + mt * 16 + j + half * 8;
        const float v = acc[mt][t][j] + bval;
        if (mode == 0) {
          ((float*)Cout)[(size_t)row * N + col] = v;
        } else if (mode == 1) {
          ((__bf16*)Cout)[(size_t)row * N + col] = (__bf16)v;
        } else {
          // Vt layout: (B, KVD, S); row = b*S + s
          const int b = row >> 11;       // / S_
          const int s = row & (S_ - 1);  // % S_
          ((__bf16*)Cout)[(size_t)b * ((size_t)KVD_ * S_) +
                          (size_t)col * S_ + s] = (__bf16)v;
        }
      }
    }
  }
}

// ---------------------------------------------------------------------------
// Flash attention: one wave per 16-row Q tile of one (b,h). Causal + ALiBi,
// online softmax, bf16 WMMA for QK^T and PV, f32 running accumulators.
// Schedule per 32-key step:
//   QK WMMAs -> issue V loads -> softmax (VALU overlaps V loads)
//   -> P via LDS -> issue next K loads -> O*corr (overlaps K loads) -> PV WMMAs
__global__ void attn_kernel(const __bf16* __restrict__ Q,
                            const __bf16* __restrict__ Kb,
                            const __bf16* __restrict__ Vt,
                            const float* __restrict__ slopes,
                            __bf16* __restrict__ O) {
  __shared__ __align__(32) __bf16 plds[4][16 * 32]; // per-wave P staging

  const int lane = threadIdx.x & 31;
  const int wave = threadIdx.x >> 5;
  const int half = lane >> 4;
  const int lr = lane & 15;

  const int tile = blockIdx.x * 4 + wave;
  const int qtiles = S_ / 16;                  // 128
  const int b = tile / (H_ * qtiles);
  const int rem = tile % (H_ * qtiles);
  const int h = rem / qtiles;
  const int q0 = (rem % qtiles) * 16;
  const int kv = h >> 2;                       // h / n_rep
  const float slope = slopes[h];

  // Preload Q tile (16 x 128) in A-layout registers.
  const __bf16* qrow = Q + (size_t)((size_t)b * S_ + q0 + lr) * D_ + h * DH_;
  v16bf qa[4];
#pragma unroll
  for (int d = 0; d < 4; ++d) qa[d] = load_a_tile(qrow + d * 32, half);

  v8f oacc[8] = {};
  float m[8], l[8];
#pragma unroll
  for (int j = 0; j < 8; ++j) { m[j] = -3.0e38f; l[j] = 0.0f; }

  __bf16* pw = &plds[wave][0];
  const __bf16* kbase = Kb + (size_t)((size_t)b * S_ + lr) * KVD_ + kv * DH_;
  const __bf16* vbase = Vt + ((size_t)b * KV_ + kv) * ((size_t)DH_ * S_);

  // kbuf[d][colhalf]: B operands for the two 16-key column halves
  v16bf kbuf[4][2];
#pragma unroll
  for (int d = 0; d < 4; ++d) {
    kbuf[d][0] = load_b_tile(kbase + d * 32, half);
    kbuf[d][1] = load_b_tile(kbase + (size_t)16 * KVD_ + d * 32, half);
  }

  const int kend = q0 + 16;
  for (int kb0 = 0; kb0 < kend; kb0 += 32) {
    // ---- S = Q K^T over 32 keys (two 16-col halves) ----
    v8f s0 = {}, s1 = {};
#pragma unroll
    for (int d = 0; d < 4; ++d) {
      s0 = wmma_bf16(qa[d], kbuf[d][0], s0);
      s1 = wmma_bf16(qa[d], kbuf[d][1], s1);
    }

    // ---- issue all V-tile loads now; softmax VALU hides their latency ----
    v16bf vbuf[8];
#pragma unroll
    for (int c = 0; c < 8; ++c)
      vbuf[c] = load_b_tile(vbase + (size_t)(c * 16 + lr) * S_ + kb0, half);

    // ---- scale + ALiBi + causal mask + online softmax ----
    const int c0 = kb0 + lr;
    const int c1 = c0 + 16;
    float p0[8], p1[8], corr[8];
#pragma unroll
    for (int j = 0; j < 8; ++j) {
      const int rq = q0 + j + half * 8;
      float v0 = (c0 <= rq) ? s0[j] * SCALE_ + slope * (float)(c0 - rq) : -3.0e38f;
      float v1 = (c1 <= rq) ? s1[j] * SCALE_ + slope * (float)(c1 - rq) : -3.0e38f;
      float mnew = fmaxf(m[j], rmax16(fmaxf(v0, v1)));
      p0[j] = __expf(v0 - mnew);
      p1[j] = __expf(v1 - mnew);
      const float rs = rsum16(p0[j] + p1[j]);
      corr[j] = __expf(m[j] - mnew);
      l[j] = l[j] * corr[j] + rs;
      m[j] = mnew;
    }

    // ---- C-layout -> LDS -> A-layout for P (wave-private, DS-ordered) ----
#pragma unroll
    for (int j = 0; j < 8; ++j) {
      const int row = j + half * 8;
      pw[row * 32 + lr] = (__bf16)p0[j];
      pw[row * 32 + 16 + lr] = (__bf16)p1[j];
    }
    asm volatile("s_wait_dscnt 0x0" ::: "memory");
    v16bf pa = load_a_tile(pw + lr * 32, half);

    // ---- issue next iteration's K-tile loads (WAR reuse of kbuf) ----
    const int knx = (kb0 + 32 < kend) ? kb0 + 32 : kb0;
#pragma unroll
    for (int d = 0; d < 4; ++d) {
      kbuf[d][0] = load_b_tile(kbase + (size_t)knx * KVD_ + d * 32, half);
      kbuf[d][1] =
          load_b_tile(kbase + (size_t)(knx + 16) * KVD_ + d * 32, half);
    }

    // ---- O = O*corr + P @ V (8 dh chunks of 16); VALU overlaps K loads ----
#pragma unroll
    for (int c = 0; c < 8; ++c) {
#pragma unroll
      for (int j = 0; j < 8; ++j) oacc[c][j] *= corr[j];
      oacc[c] = wmma_bf16(pa, vbuf[c], oacc[c]);
    }
  }

  // ---- epilogue: O /= l, store bf16 into (B,S,H,DH) == (B,S,D) ----
  float invl[8];
#pragma unroll
  for (int j = 0; j < 8; ++j) invl[j] = 1.0f / l[j];
#pragma unroll
  for (int c = 0; c < 8; ++c) {
#pragma unroll
    for (int j = 0; j < 8; ++j) {
      const int q = q0 + j + half * 8;
      O[(size_t)((size_t)b * S_ + q) * D_ + h * DH_ + c * 16 + lr] =
          (__bf16)(oacc[c][j] * invl[j]);
    }
  }
}

// ---------------------------------------------------------------------------
extern "C" void kernel_launch(void* const* d_in, const int* in_sizes, int n_in,
                              void* d_out, int out_size, void* d_ws,
                              size_t ws_size, hipStream_t stream) {
  (void)in_sizes; (void)n_in; (void)out_size; (void)ws_size;

  const float* X      = (const float*)d_in[0];
  // d_in[1] = attention_mask (pure causal -1e9; handled analytically)
  const float* Wq     = (const float*)d_in[2];
  const float* bq     = (const float*)d_in[3];
  const float* Wk     = (const float*)d_in[4];
  const float* bk     = (const float*)d_in[5];
  const float* Wv     = (const float*)d_in[6];
  const float* bv     = (const float*)d_in[7];
  const float* Wo     = (const float*)d_in[8];
  const float* bo     = (const float*)d_in[9];
  const float* slopes = (const float*)d_in[10];
  float* out = (float*)d_out;

  constexpr size_t XE  = ROWS_ * (size_t)D_;      // 8388608
  constexpr size_t WQE = (size_t)D_ * D_;         // 4194304
  constexpr size_t WKE = (size_t)KVD_ * D_;       // 1048576
  constexpr size_t KVE = ROWS_ * (size_t)KVD_;    // 2097152

  __bf16* w = (__bf16*)d_ws;
  __bf16* Xbf  = w;  w += XE;
  __bf16* Wqbf = w;  w += WQE;
  __bf16* Wkbf = w;  w += WKE;
  __bf16* Wvbf = w;  w += WKE;
  __bf16* Wobf = w;  w += WQE;
  __bf16* Qbf  = w;  w += XE;
  __bf16* Kbf  = w;  w += KVE;
  __bf16* Vtbf = w;  w += KVE;   // (B, KVD, S)
  __bf16* Obf  = w;  w += XE;

  // 1) convert inputs to bf16
  f32_to_bf16_kernel<<<4096, 256, 0, stream>>>(X,  Xbf,  XE);
  f32_to_bf16_kernel<<<4096, 256, 0, stream>>>(Wq, Wqbf, WQE);
  f32_to_bf16_kernel<<<2048, 256, 0, stream>>>(Wk, Wkbf, WKE);
  f32_to_bf16_kernel<<<2048, 256, 0, stream>>>(Wv, Wvbf, WKE);
  f32_to_bf16_kernel<<<4096, 256, 0, stream>>>(Wo, Wobf, WQE);

  const int Mrows = (int)ROWS_; // 4096

  // 2) projections
  gemm_bf16_kernel<<<dim3(Mrows / 32, D_ / 256), 128, 0, stream>>>(
      Xbf, Wqbf, bq, Qbf, Mrows, D_, D_, /*mode=*/1);
  gemm_bf16_kernel<<<dim3(Mrows / 32, KVD_ / 256), 128, 0, stream>>>(
      Xbf, Wkbf, bk, Kbf, Mrows, KVD_, D_, /*mode=*/1);
  gemm_bf16_kernel<<<dim3(Mrows / 32, KVD_ / 256), 128, 0, stream>>>(
      Xbf, Wvbf, bv, Vtbf, Mrows, KVD_, D_, /*mode=*/2);

  // 3) flash attention (4096 q-tiles, 4 waves/block)
  attn_kernel<<<(B_ * H_ * (S_ / 16)) / 4, 128, 0, stream>>>(
      Qbf, Kbf, Vtbf, slopes, Obf);

  // 4) output projection -> f32 d_out
  gemm_bf16_kernel<<<dim3(Mrows / 32, D_ / 256), 128, 0, stream>>>(
      Obf, Wobf, bo, out, Mrows, D_, D_, /*mode=*/0);
}
